// GraphNeuralNetwork_33792802685454
// MI455X (gfx1250) — matrix operand order repaired and checked by reading, not compiled, not run on hardware
//
#include <hip/hip_runtime.h>
#include <hip/hip_bf16.h>

typedef __attribute__((ext_vector_type(16))) __bf16 v16bf;
typedef __attribute__((ext_vector_type(8)))  float  v8f;

// -------- helpers --------
static __device__ __forceinline__ float sigmoid_(float x) { return 1.0f / (1.0f + expf(-x)); }
static __device__ __forceinline__ float lrelu_(float x) { return x > 0.f ? x : 0.2f * x; }
// monotone float<->uint map for atomicMax on floats
static __device__ __forceinline__ unsigned fenc(float f) {
  unsigned u = __float_as_uint(f);
  return (u & 0x80000000u) ? ~u : (u | 0x80000000u);
}
static __device__ __forceinline__ float fdec(unsigned u) {
  return __uint_as_float((u & 0x80000000u) ? (u & 0x7FFFFFFFu) : ~u);
}

// -------- WMMA GEMM: C[M,Nout] = H[M,K] @ W[K,Nout], bf16 compute, f32 accum --------
// Block = 256 threads = 8 waves. Block tile: 128 rows x 64 cols.
// W tile (32 x 64) staged in LDS as bf16, transposed to n-major so each lane's
// B fragment is 16 contiguous bf16 (32B ds load). Each wave holds one A fragment
// and 4 accumulators -> 4 v_wmma per K-step.
// Requires: M % 128 == 0, K % 32 == 0, Nout % 64 == 0 (true for all calls here).
__global__ __launch_bounds__(256) void gemm_bf16_wmma(
    const float* __restrict__ Hm, const float* __restrict__ Wm,
    float* __restrict__ Cm, int M, int K, int Nout) {
  __shared__ __bf16 lwb[64 * 32];          // [n][k] n-major, 4KB
  const int tid  = threadIdx.x;
  const int wave = tid >> 5;
  const int lane = tid & 31;
  const int stripsN = Nout >> 6;           // 64-col strips
  const int blockM  = blockIdx.x / stripsN;
  const int blockN  = blockIdx.x - blockM * stripsN;
  const int n0      = blockN << 6;
  const int rowBase = (blockM * 8 + wave) << 4;   // this wave's 16 rows
  const int m  = lane & 15;
  const int hi = lane >> 4;
  const float* hrow = Hm + (size_t)(rowBase + m) * K;

  v8f acc[4] = {};
  for (int k0 = 0; k0 < K; k0 += 32) {
    __syncthreads();
    // cooperative stage: W[k0:k0+32, n0:n0+64) -> LDS bf16 transposed
    for (int e = tid; e < 32 * 64; e += 256) {
      int kk = e >> 6, nn = e & 63;
      lwb[nn * 32 + kk] = (__bf16)Wm[(size_t)(k0 + kk) * Nout + (n0 + nn)];
    }
    __syncthreads();
    // A fragment (contiguous k, matching B): elems 0..15 -> K=k0+hi*16+i
    const float4 a0 = *reinterpret_cast<const float4*>(hrow + k0 + hi * 16);
    const float4 a1 = *reinterpret_cast<const float4*>(hrow + k0 + hi * 16 + 4);
    const float4 a2 = *reinterpret_cast<const float4*>(hrow + k0 + hi * 16 + 8);
    const float4 a3 = *reinterpret_cast<const float4*>(hrow + k0 + hi * 16 + 12);
    v16bf a;
    a[0]  = (__bf16)a0.x; a[1]  = (__bf16)a0.y; a[2]  = (__bf16)a0.z; a[3]  = (__bf16)a0.w;
    a[4]  = (__bf16)a1.x; a[5]  = (__bf16)a1.y; a[6]  = (__bf16)a1.z; a[7]  = (__bf16)a1.w;
    a[8]  = (__bf16)a2.x; a[9]  = (__bf16)a2.y; a[10] = (__bf16)a2.z; a[11] = (__bf16)a2.w;
    a[12] = (__bf16)a3.x; a[13] = (__bf16)a3.y; a[14] = (__bf16)a3.z; a[15] = (__bf16)a3.w;
#pragma unroll
    for (int j = 0; j < 4; ++j) {
      // B fragment: lanes 0-15: K=k0+0..15, lanes 16-31: K=k0+16..31; col = j*16 + m
      const v16bf b = *reinterpret_cast<const v16bf*>(&lwb[((j << 4) + m) * 32 + (hi << 4)]);
      acc[j] = __builtin_amdgcn_wmma_f32_16x16x32_bf16(false, a, false, b,
                                                       (short)0, acc[j], false, false);
    }
  }
  // C f32 16x16 layout: lane col = lane&15 ; VGPR j2 -> row = rowBase + j2 + 8*hi
#pragma unroll
  for (int j = 0; j < 4; ++j) {
    float* cp = Cm + (size_t)(rowBase + hi * 8) * Nout + n0 + (j << 4) + m;
#pragma unroll
    for (int j2 = 0; j2 < 8; ++j2) cp[(size_t)j2 * Nout] = acc[j][j2];
  }
}

// -------- degree / norm --------
__global__ void k_deg_init(float* d, int n) {
  int i = blockIdx.x * blockDim.x + threadIdx.x;
  if (i < n) d[i] = 1.0f;   // self-loop counted
}
__global__ void k_deg_count(const int* __restrict__ dst, int E, float* d) {
  int i = blockIdx.x * blockDim.x + threadIdx.x;
  if (i < E) atomicAdd(&d[dst[i]], 1.0f);
}
__global__ void k_deg_fin(float* d, int n) {
  int i = blockIdx.x * blockDim.x + threadIdx.x;
  if (i < n) d[i] = rsqrtf(d[i]);
}

// -------- GCN aggregate --------
__global__ void k_agg_init(const float* __restrict__ A, const float* __restrict__ bias,
                           const float* __restrict__ dinv, float* __restrict__ B,
                           int total, int lgF) {
  int i = blockIdx.x * blockDim.x + threadIdx.x;
  if (i >= total) return;
  int n = i >> lgF; int f = i & ((1 << lgF) - 1);
  float dv = dinv[n];
  B[i] = bias[f] + A[i] * dv * dv;   // bias + self-loop message
}
__global__ void k_agg_edges(const float* __restrict__ A, const float* __restrict__ dinv,
                            const int* __restrict__ src, const int* __restrict__ dst,
                            float* B, int total, int lgQ, int F) {
  int i = blockIdx.x * blockDim.x + threadIdx.x;
  if (i >= total) return;
  int e = i >> lgQ; int q = (i & ((1 << lgQ) - 1)) << 2;
  int s = src[e], d = dst[e];
  float w = dinv[s] * dinv[d];
  const float4 g = *reinterpret_cast<const float4*>(A + (size_t)s * F + q);
  float* out = B + (size_t)d * F + q;
  atomicAdd(out + 0, g.x * w); atomicAdd(out + 1, g.y * w);
  atomicAdd(out + 2, g.z * w); atomicAdd(out + 3, g.w * w);
}
__global__ void k_relu(float* B, int total) {
  int i = blockIdx.x * blockDim.x + threadIdx.x;
  if (i < total) B[i] = fmaxf(B[i], 0.0f);
}

// -------- GAT --------
__global__ void k_att_scores(const float* __restrict__ G, const float* __restrict__ att_s,
                             const float* __restrict__ att_d,
                             float* as_, float* ad_, int total) {
  int i = blockIdx.x * blockDim.x + threadIdx.x;
  if (i >= total) return;
  int n = i >> 2, h = i & 3;
  const float* g  = G + (size_t)n * 256 + h * 64;
  const float* vs = att_s + h * 64;
  const float* vd = att_d + h * 64;
  float s = 0.f, d = 0.f;
#pragma unroll 8
  for (int k = 0; k < 64; ++k) { float gv = g[k]; s += gv * vs[k]; d += gv * vd[k]; }
  as_[i] = s; ad_[i] = d;
}
__global__ void k_m_init(const float* as_, const float* ad_, unsigned* mu, int total) {
  int i = blockIdx.x * blockDim.x + threadIdx.x;
  if (i < total) mu[i] = fenc(lrelu_(as_[i] + ad_[i]));   // self-loop alpha
}
__global__ void k_edge_alpha(const int* __restrict__ src, const int* __restrict__ dst,
                             const float* __restrict__ as_, const float* __restrict__ ad_,
                             float* alpha, unsigned* mu, int total) {
  int i = blockIdx.x * blockDim.x + threadIdx.x;
  if (i >= total) return;
  int e = i >> 2, h = i & 3;
  int s = src[e], d = dst[e];
  float a = lrelu_(as_[s * 4 + h] + ad_[d * 4 + h]);
  alpha[i] = a;
  atomicMax(&mu[d * 4 + h], fenc(a));
}
__global__ void k_s_init(const float* as_, const float* ad_, unsigned* mu,
                         float* eself, float* ssum, int total) {
  int i = blockIdx.x * blockDim.x + threadIdx.x;
  if (i >= total) return;
  float m = fdec(mu[i]);
  ((float*)mu)[i] = m;                         // decode in place for later passes
  float es = expf(lrelu_(as_[i] + ad_[i]) - m);
  eself[i] = es; ssum[i] = es;                 // self-loop seeds the sum
}
__global__ void k_edge_exp(const int* __restrict__ dst, const float* __restrict__ mf,
                           float* alpha, float* ssum, int total) {
  int i = blockIdx.x * blockDim.x + threadIdx.x;
  if (i >= total) return;
  int e = i >> 2, h = i & 3;
  int d = dst[e];
  float ee = expf(alpha[i] - mf[d * 4 + h]);
  alpha[i] = ee;
  atomicAdd(&ssum[d * 4 + h], ee);
}
__global__ void k_gat_out_init(const float* __restrict__ G, const float* __restrict__ eself,
                               const float* __restrict__ ssum, float* B, int total) {
  int i = blockIdx.x * blockDim.x + threadIdx.x;
  if (i >= total) return;
  int n = i >> 8; int h = (i >> 6) & 3;
  B[i] = G[i] * (eself[n * 4 + h] / ssum[n * 4 + h]);   // self-loop message
}
__global__ void k_edge_msg(const int* __restrict__ src, const int* __restrict__ dst,
                           const float* __restrict__ G, const float* __restrict__ alpha,
                           const float* __restrict__ ssum, float* B, int total) {
  int i = blockIdx.x * blockDim.x + threadIdx.x;
  if (i >= total) return;
  int e = i >> 6; int c = i & 63; int h = c >> 4; int d4 = (c & 15) << 2;
  int s = src[e], d = dst[e];
  float w = alpha[e * 4 + h] / ssum[d * 4 + h];
  const float4 g = *reinterpret_cast<const float4*>(G + (size_t)s * 256 + h * 64 + d4);
  float* out = B + (size_t)d * 256 + h * 64 + d4;
  atomicAdd(out + 0, g.x * w); atomicAdd(out + 1, g.y * w);
  atomicAdd(out + 2, g.z * w); atomicAdd(out + 3, g.w * w);
}
__global__ void k_final_h(const float* __restrict__ B, const float* __restrict__ bg,
                          float* hout, int total) {
  int i = blockIdx.x * blockDim.x + threadIdx.x;
  if (i >= total) return;
  int n = i >> 6; int dd = i & 63;
  const float* r = B + (size_t)n * 256 + dd;
  hout[i] = 0.25f * (r[0] + r[64] + r[128] + r[192]) + bg[dd];   // mean over heads
}
__global__ void k_heads(const float* __restrict__ hout,
                        const float* Wc, const float* bc,
                        const float* Wd, const float* bd,
                        const float* Wp, const float* bp,
                        float* outM, float* outD, float* outP, int n) {
  int i = blockIdx.x * blockDim.x + threadIdx.x;
  if (i >= n) return;
  const float* r = hout + (size_t)i * 64;
  float c = 0.f, d = 0.f, p = 0.f;
#pragma unroll 8
  for (int k = 0; k < 64; ++k) { float v = r[k]; c += v * Wc[k]; d += v * Wd[k]; p += v * Wp[k]; }
  outM[i] = sigmoid_(c + bc[0]);
  outD[i] = sigmoid_(d + bd[0]) * 6.0f;
  outP[i] = sigmoid_(p + bp[0]);
}

// -------- host launcher --------
extern "C" void kernel_launch(void* const* d_in, const int* in_sizes, int n_in,
                              void* d_out, int out_size, void* d_ws, size_t ws_size,
                              hipStream_t stream) {
  const float* x  = (const float*)d_in[0];
  const int*   ei = (const int*)d_in[1];
  const float* W0 = (const float*)d_in[2];  const float* b0 = (const float*)d_in[3];
  const float* W1 = (const float*)d_in[4];  const float* b1 = (const float*)d_in[5];
  const float* W2 = (const float*)d_in[6];  const float* b2 = (const float*)d_in[7];
  const float* Wg = (const float*)d_in[8];
  const float* att_s = (const float*)d_in[9];
  const float* att_d = (const float*)d_in[10];
  const float* bg = (const float*)d_in[11];
  const float* Wc = (const float*)d_in[12]; const float* bc = (const float*)d_in[13];
  const float* Wd = (const float*)d_in[14]; const float* bd = (const float*)d_in[15];
  const float* Wp = (const float*)d_in[16]; const float* bp = (const float*)d_in[17];

  const int N = in_sizes[0] / 256;
  const int E = in_sizes[1] / 2;
  const int* srcA = ei;
  const int* dstA = ei + E;

  char* wsb = (char*)d_ws;
  size_t off = 0;
  auto carve = [&](size_t bytes) -> char* {
    char* p = wsb + off;
    off = (off + bytes + 255) & ~(size_t)255;
    return p;
  };
  float*    A     = (float*)carve((size_t)N * 512 * sizeof(float));  // GEMM out
  float*    Bb    = (float*)carve((size_t)N * 512 * sizeof(float));  // aggregate out
  float*    dinv  = (float*)carve((size_t)N * sizeof(float));
  float*    as_   = (float*)carve((size_t)N * 4 * sizeof(float));
  float*    ad_   = (float*)carve((size_t)N * 4 * sizeof(float));
  unsigned* mu    = (unsigned*)carve((size_t)N * 4 * sizeof(unsigned));
  float*    ssum  = (float*)carve((size_t)N * 4 * sizeof(float));
  float*    eself = (float*)carve((size_t)N * 4 * sizeof(float));
  float*    alpha = (float*)carve((size_t)E * 4 * sizeof(float));
  (void)ws_size; (void)n_in; (void)out_size;

  auto blocks = [](long long total) -> dim3 { return dim3((unsigned)((total + 255) / 256)); };
  dim3 tb(256);

  // symmetric normalization
  k_deg_init<<<blocks(N), tb, 0, stream>>>(dinv, N);
  k_deg_count<<<blocks(E), tb, 0, stream>>>(dstA, E, dinv);
  k_deg_fin<<<blocks(N), tb, 0, stream>>>(dinv, N);

  // three GCN layers
  const float* hin = x;
  const float* Ws[3] = {W0, W1, W2};
  const float* bs[3] = {b0, b1, b2};
  const int Kin[3]  = {256, 512, 256};
  const int Kout[3] = {512, 256, 128};
  const int lgF[3]  = {9, 8, 7};
  for (int l = 0; l < 3; ++l) {
    unsigned gemmBlocks = (unsigned)((N / 128) * (Kout[l] / 64));
    gemm_bf16_wmma<<<dim3(gemmBlocks), tb, 0, stream>>>(hin, Ws[l], A, N, Kin[l], Kout[l]);
    long long tot = (long long)N * Kout[l];
    k_agg_init<<<blocks(tot), tb, 0, stream>>>(A, bs[l], dinv, Bb, (int)tot, lgF[l]);
    long long etot = (long long)E * (Kout[l] / 4);
    k_agg_edges<<<blocks(etot), tb, 0, stream>>>(A, dinv, srcA, dstA, Bb, (int)etot, lgF[l] - 2, Kout[l]);
    k_relu<<<blocks(tot), tb, 0, stream>>>(Bb, (int)tot);
    hin = Bb;
  }

  // GAT: G = h3 @ Wg  -> A [N, 4*64]
  {
    unsigned gemmBlocks = (unsigned)((N / 128) * (256 / 64));
    gemm_bf16_wmma<<<dim3(gemmBlocks), tb, 0, stream>>>(Bb, Wg, A, N, 128, 256);
  }
  int nh = N * 4;
  int eh = E * 4;
  k_att_scores<<<blocks(nh), tb, 0, stream>>>(A, att_s, att_d, as_, ad_, nh);
  k_m_init<<<blocks(nh), tb, 0, stream>>>(as_, ad_, mu, nh);
  k_edge_alpha<<<blocks(eh), tb, 0, stream>>>(srcA, dstA, as_, ad_, alpha, mu, eh);
  k_s_init<<<blocks(nh), tb, 0, stream>>>(as_, ad_, mu, eself, ssum, nh);
  k_edge_exp<<<blocks(eh), tb, 0, stream>>>(dstA, (const float*)mu, alpha, ssum, eh);
  long long g256 = (long long)N * 256;
  k_gat_out_init<<<blocks(g256), tb, 0, stream>>>(A, eself, ssum, Bb, (int)g256);
  long long em = (long long)E * 64;
  k_edge_msg<<<blocks(em), tb, 0, stream>>>(srcA, dstA, A, alpha, ssum, Bb, (int)em);

  // outputs: h [N,64] | mastery [N] | difficulty [N] | prereq [N]
  float* hout = (float*)d_out;
  long long h64 = (long long)N * 64;
  k_final_h<<<blocks(h64), tb, 0, stream>>>(Bb, bg, hout, (int)h64);
  k_heads<<<blocks(N), tb, 0, stream>>>(hout, Wc, bc, Wd, bd, Wp, bp,
                                        hout + h64, hout + h64 + N, hout + h64 + 2 * N, N);
}
